// GATNet_18296560681307
// MI455X (gfx1250) — matrix operand order repaired and checked by reading, not compiled, not run on hardware
//
#include <hip/hip_runtime.h>
#include <hip/hip_bf16.h>
#include <math.h>

typedef __attribute__((ext_vector_type(8)))  _Float16 v8h;
typedef __attribute__((ext_vector_type(16))) _Float16 v16h;
typedef __attribute__((ext_vector_type(8)))  float    v8f;

#define H1  3
#define F1  114
#define C1  342     // H1*F1
#define C1P 352     // 22 tiles of 16
#define K1P 128     // K padded for gemm1
#define N1T 22
#define D2  128
#define K2P 352     // K padded for gemm2
#define N2T 8
#define NGR 64

// ---------------- helpers ----------------

static __device__ __forceinline__ float wave_sum(float v) {
  #pragma unroll
  for (int o = 16; o > 0; o >>= 1) v += __shfl_down(v, o, 32);
  return v;
}

static __device__ __forceinline__ void atomic_max_f32(float* a, float v) {
  // positive floats: int ordering matches; negative floats: uint ordering reversed
  if (v >= 0.f) atomicMax((int*)a, __float_as_int(v));
  else          atomicMin((unsigned int*)a, __float_as_uint(v));
}

// ---------------- elementwise / prep kernels ----------------

__global__ void k_fill(float* p, long n, float v) {
  long i = (long)blockIdx.x * blockDim.x + threadIdx.x;
  if (i < n) p[i] = v;
}

// dst[M,KP] (f16, K zero-padded) from src[M,K] (f32)
__global__ void k_cvt_pad(_Float16* dst, const float* src, int M, int K, int KP) {
  long i = (long)blockIdx.x * blockDim.x + threadIdx.x;
  if (i >= (long)M * KP) return;
  int m = (int)(i / KP), k = (int)(i % KP);
  dst[i] = (k < K) ? (_Float16)src[(long)m * K + k] : (_Float16)0.f;
}

// Wt[NP,KP] (f16) = transpose of W[K,Nout] (f32), zero padded in both dims
__global__ void k_transpose_pad(_Float16* dst, const float* W, int K, int Nout, int KP, int NP) {
  long i = (long)blockIdx.x * blockDim.x + threadIdx.x;
  if (i >= (long)NP * KP) return;
  int n = (int)(i / KP), k = (int)(i % KP);
  dst[i] = (n < Nout && k < K) ? (_Float16)W[(long)k * Nout + n] : (_Float16)0.f;
}

// ---------------- WMMA GEMM: C[M,ldc] = A[M,KP](f16) * Bt[ntilesN*16,KP]^T ----------------
// one wave per 16x16 output tile, K-loop in steps of 32 via v_wmma_f32_16x16x32_f16

__global__ void k_gemm_wmma(const _Float16* __restrict__ A, const _Float16* __restrict__ Bt,
                            float* __restrict__ C, int M, int KP, int ldc, int ncols,
                            int ntilesN, int totalWaves) {
  int wid = (int)((blockIdx.x * blockDim.x + threadIdx.x) >> 5);
  if (wid >= totalWaves) return;           // wave-uniform: EXEC all-ones inside
  int lane = threadIdx.x & 31;
  int tm = wid / ntilesN, tn = wid - tm * ntilesN;
  int g = lane >> 4, ml = lane & 15;

  // A frag (16-bit 16x32): lane(g,m) holds K = {8g..8g+7, 16+8g..16+8g+7} of row m
  const _Float16* arow = A  + (long)(tm * 16 + ml) * KP + 8 * g;
  // B frag (16-bit 32x16): lane(g,n) holds K = {8g..8g+7, 16+8g..16+8g+7} of column n (rows of Bt)
  const _Float16* brow = Bt + (long)(tn * 16 + ml) * KP + 8 * g;

  v8f acc = {};
  for (int k = 0; k < KP; k += 32) {
    v8h a0 = *(const v8h*)(arow + k);
    v8h a1 = *(const v8h*)(arow + k + 16);
    v8h b0 = *(const v8h*)(brow + k);
    v8h b1 = *(const v8h*)(brow + k + 16);
    v16h av, bv;
    #pragma unroll
    for (int i = 0; i < 8; ++i) { av[i] = a0[i]; av[i+8] = a1[i]; bv[i] = b0[i]; bv[i+8] = b1[i]; }
    acc = __builtin_amdgcn_wmma_f32_16x16x32_f16(false, av, false, bv, (short)0, acc,
                                                 false, false);
  }
  // D layout: VGPR r, lane l -> (M = 8*(l>>4)+r, N = l&15)
  int col = tn * 16 + ml;
  if (col < ncols) {
    #pragma unroll
    for (int r = 0; r < 8; ++r) {
      int row = tm * 16 + 8 * g + r;
      if (row < M) C[(long)row * ldc + col] = acc[r];
    }
  }
}

// ---------------- attention coefficients ----------------

// layer 1: wave per node, 3 heads of 114
__global__ void k_alpha1(const float* __restrict__ h1, const float* __restrict__ a_src,
                         const float* __restrict__ a_dst, float* as, float* ad, int N) {
  int wid = (int)((blockIdx.x * blockDim.x + threadIdx.x) >> 5);
  if (wid >= N) return;
  int lane = threadIdx.x & 31;
  const float* row = h1 + (long)wid * C1;
  #pragma unroll
  for (int h = 0; h < H1; ++h) {
    float ss = 0.f, sd = 0.f;
    for (int d = lane; d < F1; d += 32) {
      float x = row[h * F1 + d];
      ss += x * a_src[h * F1 + d];
      sd += x * a_dst[h * F1 + d];
    }
    ss = wave_sum(ss); sd = wave_sum(sd);
    if (lane == 0) { as[wid * H1 + h] = ss; ad[wid * H1 + h] = sd; }
  }
}

// layer 2: wave per node, 1 head of 128
__global__ void k_alpha2(const float* __restrict__ h2, const float* __restrict__ a_src,
                         const float* __restrict__ a_dst, float* as, float* ad, int N) {
  int wid = (int)((blockIdx.x * blockDim.x + threadIdx.x) >> 5);
  if (wid >= N) return;
  int lane = threadIdx.x & 31;
  const float* row = h2 + (long)wid * D2;
  float ss = 0.f, sd = 0.f;
  for (int d = lane; d < D2; d += 32) {
    float x = row[d];
    ss += x * a_src[d];
    sd += x * a_dst[d];
  }
  ss = wave_sum(ss); sd = wave_sum(sd);
  if (lane == 0) { as[wid] = ss; ad[wid] = sd; }
}

// ---------------- edge passes (segment softmax via atomics) ----------------

__global__ void k_edge_max(const int* __restrict__ src, const int* __restrict__ dst,
                           const float* __restrict__ as, const float* __restrict__ ad,
                           float* m, long total, int H) {
  long i = (long)blockIdx.x * blockDim.x + threadIdx.x;
  if (i >= total) return;
  int e = (int)(i / H), h = (int)(i % H);
  int s = src[e], d = dst[e];
  float v = as[s * H + h] + ad[d * H + h];
  v = v > 0.f ? v : 0.2f * v;                 // leaky_relu(0.2)
  atomic_max_f32(&m[d * H + h], v);
}

__global__ void k_edge_sum(const int* __restrict__ src, const int* __restrict__ dst,
                           const float* __restrict__ as, const float* __restrict__ ad,
                           const float* __restrict__ m, float* ssum, float* ex,
                           long total, int H) {
  long i = (long)blockIdx.x * blockDim.x + threadIdx.x;
  if (i >= total) return;
  int e = (int)(i / H), h = (int)(i % H);
  int s = src[e], d = dst[e];
  float v = as[s * H + h] + ad[d * H + h];
  v = v > 0.f ? v : 0.2f * v;
  float w = expf(v - m[d * H + h]);
  ex[i] = w;
  atomicAdd(&ssum[d * H + h], w);
}

// agg[dst, c] += ex[e, h(c)] * feat[src, c]   (thread per edge x channel)
__global__ void k_edge_agg(const int* __restrict__ src, const int* __restrict__ dst,
                           const float* __restrict__ ex, const float* __restrict__ feat,
                           float* agg, long total, int C, int Dhead, int H) {
  long i = (long)blockIdx.x * blockDim.x + threadIdx.x;
  if (i >= total) return;
  int e = (int)(i / C), c = (int)(i % C);
  int h = c / Dhead;
  int s = src[e], d = dst[e];
  atomicAdd(&agg[(long)d * C + c], ex[(long)e * H + h] * feat[(long)s * C + c]);
}

// ---------------- node finalizers ----------------

// layer1 output: elu(agg/s + b1), converted to padded f16 for gemm2
__global__ void k_finalize1(const float* __restrict__ agg, const float* __restrict__ ssum,
                            const float* __restrict__ b, _Float16* h2in, int N) {
  long i = (long)blockIdx.x * blockDim.x + threadIdx.x;
  if (i >= (long)N * C1P) return;
  int n = (int)(i / C1P), c = (int)(i % C1P);
  _Float16 o = (_Float16)0.f;
  if (c < C1) {
    float v = agg[(long)n * C1 + c] / ssum[n * H1 + c / F1] + b[c];
    v = v > 0.f ? v : (expf(v) - 1.f);        // elu
    o = (_Float16)v;
  }
  h2in[i] = o;
}

// layer2 output: relu(agg/s + b2), in place
__global__ void k_finalize2(float* agg, const float* __restrict__ ssum,
                            const float* __restrict__ b, int N) {
  long i = (long)blockIdx.x * blockDim.x + threadIdx.x;
  if (i >= (long)N * D2) return;
  int n = (int)(i / D2), c = (int)(i % D2);
  float v = agg[i] / ssum[n] + b[c];
  agg[i] = v > 0.f ? v : 0.f;
}

// ---------------- global max pool + MLP head ----------------

__global__ void k_pool(const float* __restrict__ feat, const int* __restrict__ batch,
                       float* gbuf, int N) {
  long i = (long)blockIdx.x * blockDim.x + threadIdx.x;
  if (i >= (long)N * D2) return;
  int n = (int)(i / D2), c = (int)(i % D2);
  atomic_max_f32(&gbuf[batch[n] * D2 + c], feat[i]);
}

__global__ void k_mlp(const float* __restrict__ gbuf,
                      const float* __restrict__ Wg,  const float* __restrict__ bg,
                      const float* __restrict__ Wf1, const float* __restrict__ bf1,
                      const float* __restrict__ Wf2, const float* __restrict__ bf2,
                      const float* __restrict__ Wo,  const float* __restrict__ bo,
                      float* out) {
  int r = threadIdx.x;
  if (r >= NGR) return;
  const float* grow = gbuf + r * D2;
  float t1[64];
  for (int j = 0; j < 64; ++j) {
    float acc = bg[j];
    for (int k = 0; k < D2; ++k) acc += grow[k] * Wg[k * 64 + j];
    t1[j] = acc > 0.f ? acc : 0.f;
  }
  float t2[32];
  for (int j = 0; j < 32; ++j) {
    float acc = bf1[j];
    for (int k = 0; k < 64; ++k) acc += t1[k] * Wf1[k * 32 + j];
    t2[j] = acc > 0.f ? acc : 0.f;
  }
  float t3[16];
  for (int j = 0; j < 16; ++j) {
    float acc = bf2[j];
    for (int k = 0; k < 32; ++k) acc += t2[k] * Wf2[k * 16 + j];
    t3[j] = acc > 0.f ? acc : 0.f;
  }
  float acc = bo[0];
  for (int k = 0; k < 16; ++k) acc += t3[k] * Wo[k];
  out[r] = acc;
}

// ---------------- host orchestration ----------------

extern "C" void kernel_launch(void* const* d_in, const int* in_sizes, int n_in,
                              void* d_out, int out_size, void* d_ws, size_t ws_size,
                              hipStream_t stream) {
  const float* x      = (const float*)d_in[0];
  const int*   eidx   = (const int*)  d_in[1];
  const int*   batch  = (const int*)  d_in[2];
  const float* W1     = (const float*)d_in[3];
  const float* a_src1 = (const float*)d_in[4];
  const float* a_dst1 = (const float*)d_in[5];
  const float* b1     = (const float*)d_in[6];
  const float* W2     = (const float*)d_in[7];
  const float* a_src2 = (const float*)d_in[8];
  const float* a_dst2 = (const float*)d_in[9];
  const float* b2     = (const float*)d_in[10];
  const float* Wg     = (const float*)d_in[11];
  const float* bg     = (const float*)d_in[12];
  const float* Wf1    = (const float*)d_in[13];
  const float* bf1    = (const float*)d_in[14];
  const float* Wf2    = (const float*)d_in[15];
  const float* bf2    = (const float*)d_in[16];
  const float* Wo     = (const float*)d_in[17];
  const float* bo     = (const float*)d_in[18];
  float* out = (float*)d_out;

  const int N  = in_sizes[2];       // 20000 nodes
  const int Et = in_sizes[1] / 2;   // 660000 edges incl self loops
  const int* esrc = eidx;
  const int* edst = eidx + Et;

  size_t off = 0;
  auto alloc = [&](size_t bytes) -> void* {
    void* p = (char*)d_ws + off;
    off += (bytes + 255) & ~(size_t)255;
    return p;
  };
  _Float16* xh   = (_Float16*)alloc((size_t)N * K1P * 2);
  _Float16* W1t  = (_Float16*)alloc((size_t)C1P * K1P * 2);
  float*    h1   = (float*)   alloc((size_t)N * C1 * 4);
  float*    as1  = (float*)   alloc((size_t)N * H1 * 4);
  float*    ad1  = (float*)   alloc((size_t)N * H1 * 4);
  float*    m1   = (float*)   alloc((size_t)N * H1 * 4);
  float*    s1   = (float*)   alloc((size_t)N * H1 * 4);
  float*    ex1  = (float*)   alloc((size_t)Et * H1 * 4);
  float*    agg1 = (float*)   alloc((size_t)N * C1 * 4);
  _Float16* h2in = (_Float16*)alloc((size_t)N * K2P * 2);
  _Float16* W2t  = (_Float16*)alloc((size_t)(N2T * 16) * K2P * 2);
  float*    h2   = (float*)   alloc((size_t)N * D2 * 4);
  float*    as2  = (float*)   alloc((size_t)N * 4);
  float*    ad2  = (float*)   alloc((size_t)N * 4);
  float*    m2   = (float*)   alloc((size_t)N * 4);
  float*    s2   = (float*)   alloc((size_t)N * 4);
  float*    ex2  = (float*)   alloc((size_t)Et * 4);
  float*    agg2 = (float*)   alloc((size_t)N * D2 * 4);
  float*    gbuf = (float*)   alloc((size_t)NGR * D2 * 4);
  (void)ws_size; (void)n_in; (void)out_size;

  const int TB = 256;
  auto nblk = [&](long n) { return (int)((n + TB - 1) / TB); };

  // ---- layer 1: GEMM (WMMA) ----
  k_cvt_pad      <<<nblk((long)N * K1P),   TB, 0, stream>>>(xh,  x,  N,  F1, K1P);
  k_transpose_pad<<<nblk((long)C1P * K1P), TB, 0, stream>>>(W1t, W1, F1, C1, K1P, C1P);
  int mt  = (N + 15) / 16;
  int tw1 = mt * N1T;
  k_gemm_wmma<<<nblk((long)tw1 * 32), TB, 0, stream>>>(xh, W1t, h1, N, K1P, C1, C1, N1T, tw1);

  // ---- layer 1: attention + segment softmax + aggregate ----
  k_alpha1<<<nblk((long)N * 32), TB, 0, stream>>>(h1, a_src1, a_dst1, as1, ad1, N);
  k_fill<<<nblk((long)N * H1), TB, 0, stream>>>(m1,   (long)N * H1, -INFINITY);
  k_fill<<<nblk((long)N * H1), TB, 0, stream>>>(s1,   (long)N * H1, 0.f);
  k_fill<<<nblk((long)N * C1), TB, 0, stream>>>(agg1, (long)N * C1, 0.f);
  k_edge_max<<<nblk((long)Et * H1), TB, 0, stream>>>(esrc, edst, as1, ad1, m1, (long)Et * H1, H1);
  k_edge_sum<<<nblk((long)Et * H1), TB, 0, stream>>>(esrc, edst, as1, ad1, m1, s1, ex1,
                                                     (long)Et * H1, H1);
  k_edge_agg<<<nblk((long)Et * C1), TB, 0, stream>>>(esrc, edst, ex1, h1, agg1,
                                                     (long)Et * C1, C1, F1, H1);
  k_finalize1<<<nblk((long)N * C1P), TB, 0, stream>>>(agg1, s1, b1, h2in, N);

  // ---- layer 2: GEMM (WMMA) ----
  k_transpose_pad<<<nblk((long)(N2T * 16) * K2P), TB, 0, stream>>>(W2t, W2, C1, D2, K2P, N2T * 16);
  int tw2 = mt * N2T;
  k_gemm_wmma<<<nblk((long)tw2 * 32), TB, 0, stream>>>(h2in, W2t, h2, N, K2P, D2, D2, N2T, tw2);

  // ---- layer 2: attention + segment softmax + aggregate ----
  k_alpha2<<<nblk((long)N * 32), TB, 0, stream>>>(h2, a_src2, a_dst2, as2, ad2, N);
  k_fill<<<nblk(N), TB, 0, stream>>>(m2, N, -INFINITY);
  k_fill<<<nblk(N), TB, 0, stream>>>(s2, N, 0.f);
  k_fill<<<nblk((long)N * D2), TB, 0, stream>>>(agg2, (long)N * D2, 0.f);
  k_edge_max<<<nblk(Et), TB, 0, stream>>>(esrc, edst, as2, ad2, m2, (long)Et, 1);
  k_edge_sum<<<nblk(Et), TB, 0, stream>>>(esrc, edst, as2, ad2, m2, s2, ex2, (long)Et, 1);
  k_edge_agg<<<nblk((long)Et * D2), TB, 0, stream>>>(esrc, edst, ex2, h2, agg2,
                                                     (long)Et * D2, D2, D2, 1);
  k_finalize2<<<nblk((long)N * D2), TB, 0, stream>>>(agg2, s2, b2, N);

  // ---- pool + MLP head ----
  k_fill<<<nblk(NGR * D2), TB, 0, stream>>>(gbuf, NGR * D2, -INFINITY);
  k_pool<<<nblk((long)N * D2), TB, 0, stream>>>(agg2, batch, gbuf, N);
  k_mlp<<<1, 64, 0, stream>>>(gbuf, Wg, bg, Wf1, bf1, Wf2, bf2, Wo, bo, out);
}
